// NNPred_77790447665913
// MI455X (gfx1250) — compile-verified, matrix operand order checked by reading, not run on hardware
//
#include <hip/hip_runtime.h>
#include <hip/hip_bf16.h>
#include <math.h>

// Problem constants (from reference)
#define N_AG 4096
#define TOBS 8
#define HID  256
#define G4   1024            // 4*HID gate width
#define B2   8192            // 2*N (bil + lst batched through shared LSTM weights)
#define MT   (N_AG*TOBS)     // 32768 rows (n,t)

typedef __attribute__((ext_vector_type(16))) _Float16 v16h;
typedef __attribute__((ext_vector_type(8)))  _Float16 v8h;
typedef __attribute__((ext_vector_type(8)))  float    v8f;

// ---------------------------------------------------------------------------
// WMMA fragment load, 16-bit A/B layout per CDNA5 ISA 7.12.2:
// lanes 0-15: row = lane, K = {kb..kb+7} U {kb+16..kb+23} with kb = (lane>>4)*8
// ---------------------------------------------------------------------------
__device__ inline v16h load_frag16(const _Float16* __restrict__ base, int ld) {
  int lane = threadIdx.x & 31;
  int r    = lane & 15;
  int kb   = (lane >> 4) * 8;
  const _Float16* p = base + (size_t)r * ld + kb;
  v8h lo = *(const v8h*)(p);
  v8h hi = *(const v8h*)(p + 16);
  v16h f;
#pragma unroll
  for (int i = 0; i < 8; ++i) { f[i] = lo[i]; f[i + 8] = hi[i]; }
  return f;
}

__device__ inline v8f wmma_f16(v16h a, v16h b, v8f c) {
  return __builtin_amdgcn_wmma_f32_16x16x32_f16(
      /*neg_a=*/false, a, /*neg_b=*/false, b,
      /*c_mod=*/(short)0, c, /*reuse_a=*/false, /*reuse_b=*/false);
}

// ---------------------------------------------------------------------------
// Generic WMMA GEMM: C[M,Nn] = act( A[M,K](f16) @ W[Nn,K]^T(f16) + bias (+C) )
// Block = 256 threads = 8 waves. Each wave owns a 16x64 strip: one A fragment
// reused across 4 N-tiles (4 accumulators). Software-pipelined: iteration k+1's
// fragment loads are issued before iteration k's WMMAs, so the pre-WMMA waits
// are partial (next-iter loads stay in flight) instead of s_wait_loadcnt 0.
// Block tile 128x64. grid = (M/128, Nn/64). Weights stay resident in L2.
// ---------------------------------------------------------------------------
template <bool ACT_TANH, bool OUT_HALF, bool ACC, bool BIAS>
__global__ void wmma_gemm(const _Float16* __restrict__ A, int lda,
                          const _Float16* __restrict__ W,
                          const float* __restrict__ bias,
                          float* __restrict__ Cf, _Float16* __restrict__ Ch,
                          int ldc, int K) {
  int wave = threadIdx.x >> 5;
  int lane = threadIdx.x & 31;
  int mt = blockIdx.x * 8 + wave;     // 16-row tile index
  int n0 = blockIdx.y * 64;           // 64-col slab
  const _Float16* Ab = A + (size_t)(mt * 16) * lda;
  const _Float16* Wb = W + (size_t)n0 * K;
  v8f acc[4] = {v8f{}, v8f{}, v8f{}, v8f{}};

  // Prologue: fragments for k = 0
  v16h a  = load_frag16(Ab, lda);
  v16h b0 = load_frag16(Wb, K);
  v16h b1 = load_frag16(Wb + (size_t)16 * K, K);
  v16h b2 = load_frag16(Wb + (size_t)32 * K, K);
  v16h b3 = load_frag16(Wb + (size_t)48 * K, K);

  const int ksteps = K >> 5;
  for (int ks = 0; ks < ksteps - 1; ++ks) {
    int kn = (ks + 1) << 5;
    __builtin_prefetch(Ab + kn + 32, 0, 1);        // global_prefetch_b8
    // Issue next iteration's loads before this iteration's MMAs.
    v16h an  = load_frag16(Ab + kn, lda);
    v16h bn0 = load_frag16(Wb + kn, K);
    v16h bn1 = load_frag16(Wb + (size_t)16 * K + kn, K);
    v16h bn2 = load_frag16(Wb + (size_t)32 * K + kn, K);
    v16h bn3 = load_frag16(Wb + (size_t)48 * K + kn, K);
    acc[0] = wmma_f16(a, b0, acc[0]);
    acc[1] = wmma_f16(a, b1, acc[1]);
    acc[2] = wmma_f16(a, b2, acc[2]);
    acc[3] = wmma_f16(a, b3, acc[3]);
    a = an; b0 = bn0; b1 = bn1; b2 = bn2; b3 = bn3;
  }
  // Epilogue: last k-step
  acc[0] = wmma_f16(a, b0, acc[0]);
  acc[1] = wmma_f16(a, b1, acc[1]);
  acc[2] = wmma_f16(a, b2, acc[2]);
  acc[3] = wmma_f16(a, b3, acc[3]);

  // C/D layout: VGPR g -> (M = g + 8*(lane>>4), N = lane&15)
  int nl   = lane & 15;
  int mrow = mt * 16 + ((lane >> 4) * 8);
#pragma unroll
  for (int j = 0; j < 4; ++j) {
    int n = n0 + j * 16 + nl;
    float bs = BIAS ? bias[n] : 0.0f;
#pragma unroll
    for (int g = 0; g < 8; ++g) {
      size_t idx = (size_t)(mrow + g) * ldc + n;
      float v = acc[j][g] + bs;
      if (ACC) v += Cf[idx];
      if (ACT_TANH) v = tanhf(v);
      if (OUT_HALF) Ch[idx] = (_Float16)v; else Cf[idx] = v;
    }
  }
}

// ---------------------------------------------------------------------------
// Small helper kernels
// ---------------------------------------------------------------------------
__global__ void zero_f32(float* p, int n) {
  int i = blockIdx.x * blockDim.x + threadIdx.x;
  if (i < n) p[i] = 0.0f;
}
__global__ void zero_f16(_Float16* p, int n) {
  int i = blockIdx.x * blockDim.x + threadIdx.x;
  if (i < n) p[i] = (_Float16)0.0f;
}
__global__ void f32_to_f16(const float* __restrict__ s, _Float16* __restrict__ d, int n) {
  for (int i = blockIdx.x * blockDim.x + threadIdx.x; i < n; i += gridDim.x * blockDim.x)
    d[i] = (_Float16)s[i];
}
__global__ void bias_sum(const float* a, const float* b, float* o, int n) {
  int i = blockIdx.x * blockDim.x + threadIdx.x;
  if (i < n) o[i] = a[i] + b[i];
}

// K1: per-row P, logP, rowent.  x row n == obs flat [n*16 .. n*16+15] (raw reshape)
__global__ void k1_stats(const float* __restrict__ obs, float* __restrict__ P,
                         float* __restrict__ L, float* __restrict__ R) {
  int n = blockIdx.x * blockDim.x + threadIdx.x;
  if (n >= N_AG) return;
  float v[16], s = 0.0f;
#pragma unroll
  for (int i = 0; i < 16; ++i) { v[i] = obs[n * 16 + i] + 0.01f; s += v[i]; }
  float inv = 1.0f / s, re = 0.0f;
#pragma unroll
  for (int i = 0; i < 16; ++i) {
    float p = v[i] * inv, lp = logf(p);
    P[n * 16 + i] = p; L[n * 16 + i] = lp; re += p * lp;
  }
  R[n] = re;
}

// K2 (single block): exclusive prefix sums, then
// col_sums[j] = Sum_{i>j}(rowent[i] - P_i.logP_j) + Sum_{i<j}(rowent[j] - P_j.logP_i)
// argmin -> idx[0]=idx_a, idx[1]=idx_b
__global__ void k2_colsums(const float* __restrict__ P, const float* __restrict__ L,
                           const float* __restrict__ R, float* __restrict__ prefP,
                           float* __restrict__ prefL, float* __restrict__ prefR,
                           int* __restrict__ idxOut) {
  __shared__ float totP[16];
  __shared__ float totR_s;
  int tid = threadIdx.x;
  if (tid < 16) {
    float run = 0.0f;
    for (int j = 0; j < N_AG; ++j) { prefP[j * 16 + tid] = run; run += P[j * 16 + tid]; }
    totP[tid] = run;
  } else if (tid < 32) {
    int k = tid - 16; float run = 0.0f;
    for (int j = 0; j < N_AG; ++j) { prefL[j * 16 + k] = run; run += L[j * 16 + k]; }
  } else if (tid == 32) {
    float run = 0.0f;
    for (int j = 0; j < N_AG; ++j) { prefR[j] = run; run += R[j]; }
    totR_s = run;
  }
  __syncthreads();
  float bestv = 1e30f; int besti = 0;
  for (int j = tid; j < N_AG; j += blockDim.x) {
    float r  = R[j];
    float t1 = totR_s - prefR[j] - r;       // suffix rowent
    float t2 = (float)j * r;
#pragma unroll
    for (int k = 0; k < 16; ++k) {
      float pj = P[j * 16 + k], lj = L[j * 16 + k];
      t1 -= (totP[k] - prefP[j * 16 + k] - pj) * lj;
      t2 -= pj * prefL[j * 16 + k];
    }
    float cs = t1 + t2;
    if (cs < bestv || (cs == bestv && j < besti)) { bestv = cs; besti = j; }
  }
  __shared__ float sv[256]; __shared__ int si[256];
  sv[tid] = bestv; si[tid] = besti; __syncthreads();
  for (int s = 128; s > 0; s >>= 1) {
    if (tid < s) {
      if (sv[tid + s] < sv[tid] || (sv[tid + s] == sv[tid] && si[tid + s] < si[tid])) {
        sv[tid] = sv[tid + s]; si[tid] = si[tid + s];
      }
    }
    __syncthreads();
  }
  if (tid == 0) { idxOut[0] = si[0]; idxOut[1] = min(si[0] + 1, N_AG - 1); }
}

// K3 (1 thread): C = pinv(A)@B via (A^T A)^-1 A^T B;  Tm = C + C@fc3_W^T + fc3_b; Tinv
__global__ void k3_tm(const float* __restrict__ obs, const int* __restrict__ idx,
                      const float* __restrict__ fc3W, const float* __restrict__ fc3b,
                      float* __restrict__ TmTi) {
  if (threadIdx.x || blockIdx.x) return;
  int ia = idx[0], ib = idx[1];
  float g00 = 0, g01 = 0, g11 = 0, ab00 = 0, ab01 = 0, ab10 = 0, ab11 = 0;
  for (int t = 0; t < TOBS; ++t) {
    float a0 = obs[ia * 16 + t * 2], a1 = obs[ia * 16 + t * 2 + 1];
    float b0 = obs[ib * 16 + t * 2], b1 = obs[ib * 16 + t * 2 + 1];
    g00 += a0 * a0; g01 += a0 * a1; g11 += a1 * a1;
    ab00 += a0 * b0; ab01 += a0 * b1; ab10 += a1 * b0; ab11 += a1 * b1;
  }
  float det = g00 * g11 - g01 * g01;
  float id  = (det != 0.0f) ? 1.0f / det : 0.0f;
  float i00 = g11 * id, i01 = -g01 * id, i11 = g00 * id;
  float C00 = i00 * ab00 + i01 * ab10, C01 = i00 * ab01 + i01 * ab11;
  float C10 = i01 * ab00 + i11 * ab10, C11 = i01 * ab01 + i11 * ab11;
  float T0 = C00 + C00 * fc3W[0] + C01 * fc3W[1] + fc3b[0];
  float T1 = C01 + C00 * fc3W[2] + C01 * fc3W[3] + fc3b[1];
  float T2 = C10 + C10 * fc3W[0] + C11 * fc3W[1] + fc3b[0];
  float T3 = C11 + C10 * fc3W[2] + C11 * fc3W[3] + fc3b[1];
  float dt = T0 * T3 - T1 * T2;
  float idt = (dt != 0.0f) ? 1.0f / dt : 0.0f;
  TmTi[0] = T0; TmTi[1] = T1; TmTi[2] = T2; TmTi[3] = T3;
  TmTi[4] = T3 * idt; TmTi[5] = -T1 * idt; TmTi[6] = -T2 * idt; TmTi[7] = T0 * idt;
}

// K4: y = where(is_a, x, x@Tm); scaled = y[::-1]; write output + inp + both input
// projections (f16) as a single batched (2N, T, HID) buffer. One block per out row m.
__global__ void k4_transform(const float* __restrict__ obs, const int* __restrict__ idx,
                             const float* __restrict__ TmTi,
                             const float* __restrict__ Wb, const float* __restrict__ bb,
                             const float* __restrict__ Wl, const float* __restrict__ bl,
                             float* __restrict__ out, float* __restrict__ inp,
                             _Float16* __restrict__ proj) {
  int m = blockIdx.x, tid = threadIdx.x;
  __shared__ float y[16];
  int nsrc = N_AG - 1 - m;
  if (tid < 16) {
    int t = tid >> 1, d = tid & 1;
    float x0 = obs[nsrc * 16 + t * 2], x1 = obs[nsrc * 16 + t * 2 + 1];
    float v = (nsrc == idx[0]) ? obs[nsrc * 16 + tid]
                               : (x0 * TmTi[0 * 2 + d] + x1 * TmTi[1 * 2 + d]);
    y[tid] = v; out[m * 16 + tid] = v; inp[m * 16 + tid] = v;
  }
  __syncthreads();
  int h = tid;  // 256 threads = HID
  for (int t = 0; t < TOBS; ++t) {
    float x0 = y[t * 2], x1 = y[t * 2 + 1];
    proj[((size_t)m * TOBS + t) * HID + h] =
        (_Float16)(x0 * Wb[h * 2] + x1 * Wb[h * 2 + 1] + bb[h]);           // bilstm half
    proj[((size_t)(N_AG + m) * TOBS + t) * HID + h] =
        (_Float16)(x0 * Wl[h * 2] + x1 * Wl[h * 2 + 1] + bl[h]);           // lstm half
  }
}

// LSTM cell pointwise: gates order [i|f|g|o]; writes new c, h (f16) and h into hseq
// (hseq base pre-offset by t*HID, batch row stride TOBS*HID).
__global__ void lstm_cell(const float* __restrict__ gates, float* __restrict__ c,
                          _Float16* __restrict__ h16, _Float16* __restrict__ hseq) {
  int idx = blockIdx.x * blockDim.x + threadIdx.x;  // b*256 + h
  int b = idx >> 8, hh = idx & 255;
  const float* g = gates + (size_t)b * G4;
  float gi = g[hh], gf = g[256 + hh], gg = g[512 + hh], go = g[768 + hh];
  float cv = c[idx];
  float si = 1.0f / (1.0f + expf(-gi));
  float sf = 1.0f / (1.0f + expf(-gf));
  float so = 1.0f / (1.0f + expf(-go));
  cv = sf * cv + si * tanhf(gg);
  float hv = so * tanhf(cv);
  c[idx] = cv;
  h16[idx] = (_Float16)hv;
  hseq[(size_t)b * (TOBS * HID) + hh] = (_Float16)hv;
}

// S = lst + bil  (two halves of the batched h2seq)
__global__ void sum_halves(const _Float16* __restrict__ h2, _Float16* __restrict__ S) {
  for (size_t i = (size_t)blockIdx.x * blockDim.x + threadIdx.x; i < (size_t)MT * HID;
       i += (size_t)gridDim.x * blockDim.x)
    S[i] = (_Float16)((float)h2[i] + (float)h2[i + (size_t)MT * HID]);
}

// Head tail: out64 = O1 + inp@in2out^T + b; pred = out64@fc2^T + b2; pred_t via Tinv;
// masked squared error; uniform segments of 16 => loss = sum(l2)/ (4096*16)
__global__ void head_tail(const float* __restrict__ O1, const float* __restrict__ inp,
                          const float* __restrict__ Wio, const float* __restrict__ bio,
                          const float* __restrict__ W2, const float* __restrict__ b2,
                          const float* __restrict__ TmTi, const int* __restrict__ idx,
                          const float* __restrict__ mask, float* __restrict__ loss) {
  int row = blockIdx.x * blockDim.x + threadIdx.x;
  if (row >= MT) return;
  int n = row >> 3, t = row & 7;
  float x0 = inp[n * 16 + t * 2], x1 = inp[n * 16 + t * 2 + 1];
  float p0 = b2[0], p1 = b2[1];
  const float* o1 = O1 + (size_t)row * 64;
#pragma unroll 8
  for (int d = 0; d < 64; ++d) {
    float o = o1[d] + x0 * Wio[d * 2] + x1 * Wio[d * 2 + 1] + bio[d];
    p0 += o * W2[d];
    p1 += o * W2[64 + d];
  }
  float q0 = p0, q1 = p1;
  if (n != idx[0]) {                      // pred @ Tinv
    q0 = p0 * TmTi[4] + p1 * TmTi[6];
    q1 = p0 * TmTi[5] + p1 * TmTi[7];
  }
  float mm = mask[n * 16 + 8 + t];
  float d0 = q0 - x0, d1 = q1 - x1;
  atomicAdd(loss, mm * (d0 * d0 + d1 * d1) * (1.0f / (float)(N_AG * 16)));
}

// ---------------------------------------------------------------------------
extern "C" void kernel_launch(void* const* d_in, const int* in_sizes, int n_in,
                              void* d_out, int out_size, void* d_ws, size_t ws_size,
                              hipStream_t stream) {
  const float* obs    = (const float*)d_in[0];
  const float* lmask  = (const float*)d_in[1];
  const float* i2l_W  = (const float*)d_in[2];
  const float* i2l_b  = (const float*)d_in[3];
  const float* i2b_W  = (const float*)d_in[4];
  const float* i2b_b  = (const float*)d_in[5];
  const float* Wih0   = (const float*)d_in[6];
  const float* Whh0   = (const float*)d_in[7];
  const float* bih0   = (const float*)d_in[8];
  const float* bhh0   = (const float*)d_in[9];
  const float* Wih1   = (const float*)d_in[10];
  const float* Whh1   = (const float*)d_in[11];
  const float* bih1   = (const float*)d_in[12];
  const float* bhh1   = (const float*)d_in[13];
  const float* fc0_W  = (const float*)d_in[14];
  const float* fc0_b  = (const float*)d_in[15];
  const float* fc1_W  = (const float*)d_in[16];
  const float* fc1_b  = (const float*)d_in[17];
  const float* fc2_W  = (const float*)d_in[18];
  const float* fc2_b  = (const float*)d_in[19];
  const float* fc3_W  = (const float*)d_in[20];
  const float* fc3_b  = (const float*)d_in[21];
  const float* i2o_W  = (const float*)d_in[22];
  const float* i2o_b  = (const float*)d_in[23];

  float* out  = (float*)d_out;          // [0,65536) = scaled, [65536] = loss
  float* loss = out + 65536;

  // Workspace carve-up (~186 MB total)
  char* ws = (char*)d_ws;
  size_t off = 0;
  auto carve = [&](size_t bytes) {
    size_t o = off; off = (off + bytes + 255) & ~(size_t)255; return o;
  };
  float*    P      = (float*)(ws + carve((size_t)N_AG * 16 * 4));
  float*    Lg     = (float*)(ws + carve((size_t)N_AG * 16 * 4));
  float*    R      = (float*)(ws + carve((size_t)N_AG * 4));
  float*    prefP  = (float*)(ws + carve((size_t)N_AG * 16 * 4));
  float*    prefL  = (float*)(ws + carve((size_t)N_AG * 16 * 4));
  float*    prefR  = (float*)(ws + carve((size_t)N_AG * 4));
  int*      idx    = (int*)(ws + carve(64));
  float*    TmTi   = (float*)(ws + carve(64));
  float*    inp    = (float*)(ws + carve((size_t)N_AG * 16 * 4));
  _Float16* proj   = (_Float16*)(ws + carve((size_t)B2 * TOBS * HID * 2));   // 32MB
  _Float16* h1seq  = (_Float16*)(ws + carve((size_t)B2 * TOBS * HID * 2));   // 32MB
  _Float16* h2seq  = (_Float16*)(ws + carve((size_t)B2 * TOBS * HID * 2));   // 32MB
  _Float16* S      = (_Float16*)(ws + carve((size_t)MT * HID * 2));          // 16MB
  _Float16* O0     = (_Float16*)(ws + carve((size_t)MT * 128 * 2));          // 8MB
  float*    O1     = (float*)(ws + carve((size_t)MT * 64 * 4));              // 8MB
  float*    gates  = (float*)(ws + carve((size_t)B2 * G4 * 4));              // 32MB
  _Float16* h16    = (_Float16*)(ws + carve((size_t)B2 * HID * 2));          // 4MB
  float*    cst    = (float*)(ws + carve((size_t)B2 * HID * 4));             // 8MB
  _Float16* Wih0h  = (_Float16*)(ws + carve((size_t)G4 * HID * 2));
  _Float16* Whh0h  = (_Float16*)(ws + carve((size_t)G4 * HID * 2));
  _Float16* Wih1h  = (_Float16*)(ws + carve((size_t)G4 * HID * 2));
  _Float16* Whh1h  = (_Float16*)(ws + carve((size_t)G4 * HID * 2));
  _Float16* fc0h   = (_Float16*)(ws + carve((size_t)128 * 256 * 2));
  _Float16* fc1h   = (_Float16*)(ws + carve((size_t)64 * 128 * 2));
  float*    bcat0  = (float*)(ws + carve((size_t)G4 * 4));
  float*    bcat1  = (float*)(ws + carve((size_t)G4 * 4));
  (void)ws_size; (void)in_sizes; (void)n_in; (void)out_size;

  // ---- Phase 1: statistics / argmin / Tm / transform ----
  k1_stats<<<(N_AG + 255) / 256, 256, 0, stream>>>(obs, P, Lg, R);
  k2_colsums<<<1, 256, 0, stream>>>(P, Lg, R, prefP, prefL, prefR, idx);
  k3_tm<<<1, 1, 0, stream>>>(obs, idx, fc3_W, fc3_b, TmTi);
  k4_transform<<<N_AG, 256, 0, stream>>>(obs, idx, TmTi, i2b_W, i2b_b, i2l_W, i2l_b,
                                         out, inp, proj);

  // ---- Weight prep: f32 -> f16, bias sums ----
  f32_to_f16<<<1024, 256, 0, stream>>>(Wih0, Wih0h, G4 * HID);
  f32_to_f16<<<1024, 256, 0, stream>>>(Whh0, Whh0h, G4 * HID);
  f32_to_f16<<<1024, 256, 0, stream>>>(Wih1, Wih1h, G4 * HID);
  f32_to_f16<<<1024, 256, 0, stream>>>(Whh1, Whh1h, G4 * HID);
  f32_to_f16<<<128, 256, 0, stream>>>(fc0_W, fc0h, 128 * 256);
  f32_to_f16<<<32, 256, 0, stream>>>(fc1_W, fc1h, 64 * 128);
  bias_sum<<<4, 256, 0, stream>>>(bih0, bhh0, bcat0, G4);
  bias_sum<<<4, 256, 0, stream>>>(bih1, bhh1, bcat1, G4);

  // ---- Phase 2: 2-layer LSTM, batch 2N=8192, shared weights ----
  const int nhc = B2 * HID;
  for (int layer = 0; layer < 2; ++layer) {
    const _Float16* lin  = (layer == 0) ? proj : h1seq;
    _Float16*       hout = (layer == 0) ? h1seq : h2seq;
    const _Float16* Wih  = (layer == 0) ? Wih0h : Wih1h;
    const _Float16* Whh  = (layer == 0) ? Whh0h : Whh1h;
    const float*    bc   = (layer == 0) ? bcat0 : bcat1;
    zero_f16<<<nhc / 256, 256, 0, stream>>>(h16, nhc);
    zero_f32<<<nhc / 256, 256, 0, stream>>>(cst, nhc);
    for (int t = 0; t < TOBS; ++t) {
      // gates = x_t @ Wih^T + (bih + bhh)
      wmma_gemm<false, false, false, true>
          <<<dim3(B2 / 128, G4 / 64), 256, 0, stream>>>(
              lin + t * HID, TOBS * HID, Wih, bc, gates, nullptr, G4, HID);
      // gates += h_{t-1} @ Whh^T
      wmma_gemm<false, false, true, false>
          <<<dim3(B2 / 128, G4 / 64), 256, 0, stream>>>(
              h16, HID, Whh, nullptr, gates, nullptr, G4, HID);
      lstm_cell<<<nhc / 256, 256, 0, stream>>>(gates, cst, h16, hout + t * HID);
    }
  }

  // ---- Phase 3: head ----
  sum_halves<<<8192, 256, 0, stream>>>(h2seq, S);
  // O0 = tanh(S @ fc0^T + b0)  (f16 out)
  wmma_gemm<true, true, false, true>
      <<<dim3(MT / 128, 128 / 64), 256, 0, stream>>>(
          S, HID, fc0h, fc0_b, nullptr, O0, 128, HID);
  // O1 = tanh(O0 @ fc1^T + b1) (f32 out)
  wmma_gemm<true, false, false, true>
      <<<dim3(MT / 128, 64 / 64), 256, 0, stream>>>(
          O0, 128, fc1h, fc1_b, O1, nullptr, 64, 128);

  zero_f32<<<1, 32, 0, stream>>>(loss, 1);
  head_tail<<<MT / 256, 256, 0, stream>>>(O1, inp, i2o_W, i2o_b, fc2_W, fc2_b,
                                          TmTi, idx, lmask, loss);
}
